// DirectedNetworkFeatureExtractor_858993459727
// MI455X (gfx1250) — matrix operand order repaired and checked by reading, not compiled, not run on hardware
//
#include <hip/hip_runtime.h>
#include <hip/hip_bf16.h>

typedef _Float16 v16h __attribute__((ext_vector_type(16)));
typedef _Float16 v8h  __attribute__((ext_vector_type(8)));
typedef float    v8f  __attribute__((ext_vector_type(8)));

#define NN    100000
#define EE    1600000
#define NPAD  100096   /* ceil(NN/128)*128 */
#define NCOLS 128
#define KMAX  256
#define BPAD  8        /* LDS pad (halves): row stride (K+8)*2B stays 16B-aligned */

// ---------------------------------------------------------------------------
// ordered-float encoding for atomicMax on signed floats via unsigned atomicMax
// ---------------------------------------------------------------------------
__device__ __forceinline__ unsigned f2ord(float f) {
  unsigned u = __float_as_uint(f);
  return (u & 0x80000000u) ? ~u : (u | 0x80000000u);
}
__device__ __forceinline__ float ord2f(unsigned u) {
  return (u & 0x80000000u) ? __uint_as_float(u & 0x7fffffffu)
                           : __uint_as_float(~u);
}

// ---------------------------------------------------------------------------
// f32 -> f16 conversion (optionally a+b), zero-pads tail rows
// ---------------------------------------------------------------------------
__global__ void conv_f16_kernel(const float* __restrict__ a,
                                const float* __restrict__ b,
                                _Float16* __restrict__ out,
                                long nValid, long nTotal) {
  long i = (long)blockIdx.x * blockDim.x + threadIdx.x;
  if (i >= nTotal) return;
  float v = 0.0f;
  if (i < nValid) v = b ? (a[i] + b[i]) : a[i];
  out[i] = (_Float16)v;
}

// weight f32 [K][128] -> f16 transposed Wt[128][K]  (tiny, once per GEMM)
__global__ void conv_w_t_kernel(const float* __restrict__ W,
                                _Float16* __restrict__ Wt, int K) {
  long i = (long)blockIdx.x * blockDim.x + threadIdx.x;
  if (i >= (long)K * NCOLS) return;
  int k = (int)(i >> 7);      // row in W
  int n = (int)(i & 127);     // col in W
  Wt[(long)n * K + k] = (_Float16)W[i];
}

__global__ void zero_f32_kernel(float* __restrict__ p, long n) {
  long i = (long)blockIdx.x * blockDim.x + threadIdx.x;
  if (i < n) p[i] = 0.0f;
}

__global__ void init_gat_kernel(unsigned* __restrict__ m, float* __restrict__ s, long n) {
  long i = (long)blockIdx.x * blockDim.x + threadIdx.x;
  if (i < n) { m[i] = 0x007FFFFFu; /* f2ord(-inf) */ s[i] = 0.0f; }
}

// ---------------------------------------------------------------------------
// WMMA GEMM: out[r, colOff + c] = relu?(X[r,:K] @ W[:K,:128] + bias[c])
// Xh: f16 [NPAD][K] row-major.  Wt: f16 [128][K] (pre-transposed weight).
// Block = 256 thr (8 waves) -> 128x128 output tile.
// Entire weight (<=64KB) staged into LDS once; K loop has no barriers.
// sched_group_barrier forces: 2 VMEM reads (A) -> 16 DS reads (all B frags)
// -> 8 WMMAs per chunk, so the WMMA chain drains DS loads with partial waits.
//   kshift = log2(K)  (7 for K=128, 8 for K=256)
// ---------------------------------------------------------------------------
__global__ __launch_bounds__(256) void gemm_wmma_kernel(
    const _Float16* __restrict__ Xh, const _Float16* __restrict__ Wt,
    const float* __restrict__ bias, float* __restrict__ out,
    int K, int kshift, int ldo, int colOff, int doRelu)
{
  __shared__ _Float16 Bs[128 * (KMAX + BPAD)];   // 67.5 KB (of 320KB/WGP)

  const int t    = threadIdx.x;
  const int wid  = t >> 5;
  const int lane = t & 31;
  const int l16  = lane & 15;
  const int hi   = lane >> 4;            // 0 or 1 (lane half)
  const int ldb  = K + BPAD;
  const long rowBase = (long)blockIdx.x * 128;
  const long arow    = rowBase + wid * 16 + l16;

  // ---- stage whole pre-transposed weight into LDS: contiguous v8h copies ----
  {
    const int nChunks = (NCOLS * K) >> 3;          // v8h chunks
    const int cPerRow = K >> 3;                    // v8h chunks per LDS row
    for (int c = t; c < nChunks; c += 256) {
      int n   = c >> (kshift - 3);                 // c / cPerRow
      int off = (c - n * cPerRow) << 3;            // half offset within row
      *(v8h*)&Bs[n * ldb + off] = *(const v8h*)&Wt[(long)n * K + off];
    }
  }
  __syncthreads();

  v8f acc[8];
  const v8f vzero = {0.f,0.f,0.f,0.f,0.f,0.f,0.f,0.f};
#pragma unroll
  for (int i = 0; i < 8; ++i) acc[i] = vzero;

  const _Float16* aRow = Xh + arow * K;

  for (int k0 = 0; k0 < K; k0 += 32) {
    // A fragment (16-bit A layout: halves 0-7 -> K=Kb..Kb+7, 8-15 -> Kb+16..Kb+23)
    const _Float16* ap = aRow + k0 + hi * 8;
    if (k0 + 32 < K) __builtin_prefetch(ap + 32, 0, 3);
    v16h a;
    {
      const v8h a0 = *(const v8h*)(ap);
      const v8h a1 = *(const v8h*)(ap + 16);
#pragma unroll
      for (int i = 0; i < 8; ++i) { a[i] = a0[i]; a[i + 8] = a1[i]; }
    }

    // preload all 8 B fragments (lane = column l16, K = k0 + 8*hi + 0..7, k0 + 16 + 8*hi + 0..7)
    v16h b[8];
#pragma unroll
    for (int ct = 0; ct < 8; ++ct) {
      const _Float16* bp = &Bs[(ct * 16 + l16) * ldb + k0 + hi * 8];
      const v8h b0 = *(const v8h*)(bp);
      const v8h b1 = *(const v8h*)(bp + 16);
#pragma unroll
      for (int i = 0; i < 8; ++i) { b[ct][i] = b0[i]; b[ct][i + 8] = b1[i]; }
    }

#pragma unroll
    for (int ct = 0; ct < 8; ++ct) {
      acc[ct] = __builtin_amdgcn_wmma_f32_16x16x32_f16(
          false, a, false, b[ct], (short)0, acc[ct], false, false);
    }

    // scheduling: A vmem reads, then all 16 ds reads, then the 8 WMMAs
    __builtin_amdgcn_sched_group_barrier(0x020, 2, 0);   // VMEM read x2
    __builtin_amdgcn_sched_group_barrier(0x100, 16, 0);  // DS read  x16
    __builtin_amdgcn_sched_group_barrier(0x008, 8, 0);   // WMMA     x8
  }

  // C/D layout: lane l -> N=l%16, VGPR i -> M = i + 8*(l>=16)
  const long orow0 = rowBase + wid * 16 + hi * 8;
#pragma unroll
  for (int ct = 0; ct < 8; ++ct) {
    const int gcol = ct * 16 + l16;
    const float bv = bias ? bias[gcol] : 0.0f;
#pragma unroll
    for (int i = 0; i < 8; ++i) {
      float v = acc[ct][i] + bv;
      if (doRelu) v = v > 0.f ? v : 0.f;
      out[(orow0 + i) * (long)ldo + colOff + gcol] = v;
    }
  }
}

// ---------------------------------------------------------------------------
// GAT: attention logits  al_s[n,h] = <h[n,h,:], a_src[h,:]>, same for a_dst
// ---------------------------------------------------------------------------
__global__ void attn_logits_kernel(const float* __restrict__ hbuf,
                                   const float* __restrict__ a_src,
                                   const float* __restrict__ a_dst,
                                   float* __restrict__ al_s,
                                   float* __restrict__ al_d,
                                   int H, int C) {
  long tid = (long)blockIdx.x * blockDim.x + threadIdx.x;
  if (tid >= (long)NN * H) return;
  int n = (int)(tid / H);
  int h = (int)(tid - (long)n * H);
  const float* hp = hbuf + (long)n * 128 + h * C;
  const float* as = a_src + h * C;
  const float* ad = a_dst + h * C;
  float ss = 0.f, sd = 0.f;
  for (int c = 0; c < C; ++c) { float v = hp[c]; ss += v * as[c]; sd += v * ad[c]; }
  al_s[tid] = ss;
  al_d[tid] = sd;
}

// per-edge: e = leaky_relu(al_s[src]+al_d[dst]); segment max into m[dst]
__global__ void edge_max_kernel(const int* __restrict__ src, const int* __restrict__ dst,
                                const float* __restrict__ al_s, const float* __restrict__ al_d,
                                float* __restrict__ ebuf, unsigned* __restrict__ m, int H) {
  long e = (long)blockIdx.x * blockDim.x + threadIdx.x;
  if (e >= EE) return;
  int s = src[e], d = dst[e];
  for (int h = 0; h < H; ++h) {
    float v = al_s[(long)s * H + h] + al_d[(long)d * H + h];
    v = v > 0.f ? v : 0.2f * v;                 // leaky_relu slope 0.2
    ebuf[e * H + h] = v;
    atomicMax(&m[(long)d * H + h], f2ord(v));
  }
}

// per-edge: p = exp(e - m[dst]); segment sum into s[dst]; ebuf <- p
__global__ void edge_expsum_kernel(const int* __restrict__ dst,
                                   float* __restrict__ ebuf,
                                   const unsigned* __restrict__ m,
                                   float* __restrict__ ssum, int H) {
  long e = (long)blockIdx.x * blockDim.x + threadIdx.x;
  if (e >= EE) return;
  int d = dst[e];
  for (int h = 0; h < H; ++h) {
    float mm = ord2f(m[(long)d * H + h]);
    float p = __expf(ebuf[e * H + h] - mm);
    ebuf[e * H + h] = p;
    atomicAdd(&ssum[(long)d * H + h], p);
  }
}

// per-edge weighted scatter: agg[dst, :] += h[src, :] * alpha ; 32 lanes/edge
__global__ void scatter_kernel(const int* __restrict__ src, const int* __restrict__ dst,
                               const float* __restrict__ pbuf, const float* __restrict__ ssum,
                               const float* __restrict__ hbuf, float* __restrict__ agg,
                               int H, int shiftC) {
  long e = (long)blockIdx.x * 8 + (threadIdx.x >> 5);
  if (e >= EE) return;
  int lane = threadIdx.x & 31;
  int s = src[e], d = dst[e];
#pragma unroll
  for (int j = 0; j < 4; ++j) {
    int comp = j * 32 + lane;                   // 0..127
    int h = comp >> shiftC;                     // head index
    float alpha = pbuf[e * H + h] / (ssum[(long)d * H + h] + 1e-16f);
    atomicAdd(&agg[(long)d * 128 + comp], hbuf[(long)s * 128 + comp] * alpha);
  }
}

// x2 = relu(agg + bias) written into out[:, colOff:colOff+128]
__global__ void gat_finish_kernel(const float* __restrict__ agg,
                                  const float* __restrict__ bias,
                                  float* __restrict__ out, int ldo, int colOff) {
  long i = (long)blockIdx.x * blockDim.x + threadIdx.x;
  if (i >= (long)NPAD * 128) return;
  long row = i >> 7;
  int  c   = (int)(i & 127);
  float v = 0.f;
  if (row < NN) { v = agg[i] + bias[c]; v = v > 0.f ? v : 0.f; }
  out[row * (long)ldo + colOff + c] = v;
}

// d_out = x4 + relu(agg + fg_b)
__global__ void final_out_kernel(const float* __restrict__ x4,
                                 const float* __restrict__ agg,
                                 const float* __restrict__ fgb,
                                 float* __restrict__ out) {
  long i = (long)blockIdx.x * blockDim.x + threadIdx.x;
  if (i >= (long)NN * 128) return;
  int c = (int)(i & 127);
  float v = agg[i] + fgb[c];
  v = v > 0.f ? v : 0.f;
  out[i] = x4[i] + v;
}

// ---------------------------------------------------------------------------
// host-side orchestration
// ---------------------------------------------------------------------------
static inline unsigned gdiv(long n, int b) { return (unsigned)((n + b - 1) / b); }

struct GatBufs {
  float* als; float* ald; unsigned* m; float* ssum; float* ebuf; float* agg;
};

static void run_gemm(hipStream_t st, const _Float16* Xh, const _Float16* Wt,
                     const float* bias, float* out, int K, int ldo, int colOff, int relu) {
  int kshift = (K == 128) ? 7 : 8;
  gemm_wmma_kernel<<<NPAD / 128, 256, 0, st>>>(Xh, Wt, bias, out, K, kshift, ldo, colOff, relu);
}

static void run_conv_w(hipStream_t st, const float* W, _Float16* Wt, int K) {
  long n = (long)K * NCOLS;
  conv_w_t_kernel<<<gdiv(n, 256), 256, 0, st>>>(W, Wt, K);
}

static void run_gat(hipStream_t st, const float* hbuf, const int* src, const int* dst,
                    const float* a_s, const float* a_d, int H, int C, int shiftC,
                    const GatBufs& g) {
  attn_logits_kernel<<<gdiv((long)NN * H, 256), 256, 0, st>>>(hbuf, a_s, a_d, g.als, g.ald, H, C);
  init_gat_kernel<<<gdiv((long)NN * H, 256), 256, 0, st>>>(g.m, g.ssum, (long)NN * H);
  zero_f32_kernel<<<gdiv((long)NPAD * 128, 256), 256, 0, st>>>(g.agg, (long)NPAD * 128);
  edge_max_kernel<<<gdiv(EE, 256), 256, 0, st>>>(src, dst, g.als, g.ald, g.ebuf, g.m, H);
  edge_expsum_kernel<<<gdiv(EE, 256), 256, 0, st>>>(dst, g.ebuf, g.m, g.ssum, H);
  scatter_kernel<<<gdiv(EE, 8), 256, 0, st>>>(src, dst, g.ebuf, g.ssum, hbuf, g.agg, H, shiftC);
}

extern "C" void kernel_launch(void* const* d_in, const int* in_sizes, int n_in,
                              void* d_out, int out_size, void* d_ws, size_t ws_size,
                              hipStream_t stream) {
  (void)in_sizes; (void)n_in; (void)out_size; (void)ws_size;

  const float* x     = (const float*)d_in[0];
  const int*   ei    = (const int*)d_in[1];
  const int*   src   = ei;
  const int*   dst   = ei + EE;
  const float* fc1_W = (const float*)d_in[2];  const float* fc1_b = (const float*)d_in[3];
  const float* g1_W  = (const float*)d_in[4];  const float* g1_as = (const float*)d_in[5];
  const float* g1_ad = (const float*)d_in[6];  const float* g1_b  = (const float*)d_in[7];
  const float* mfc_W = (const float*)d_in[8];  const float* mfc_b = (const float*)d_in[9];
  const float* mg_W  = (const float*)d_in[10]; const float* mg_as = (const float*)d_in[11];
  const float* mg_ad = (const float*)d_in[12]; const float* mg_b  = (const float*)d_in[13];
  const float* ffc_W = (const float*)d_in[14]; const float* ffc_b = (const float*)d_in[15];
  const float* fg_W  = (const float*)d_in[16]; const float* fg_as = (const float*)d_in[17];
  const float* fg_ad = (const float*)d_in[18]; const float* fg_b  = (const float*)d_in[19];
  float* out = (float*)d_out;

  // ---- carve workspace ----
  char* p = (char*)d_ws;
  auto carve = [&](size_t bytes) -> void* {
    void* r = (void*)p;
    p += (bytes + 255) & ~(size_t)255;
    return r;
  };
  _Float16* Xh   = (_Float16*)carve((size_t)NPAD * 256 * 2);
  _Float16* Wt   = (_Float16*)carve((size_t)KMAX * 128 * 2);
  float* first   = (float*)carve((size_t)NPAD * 256 * 4);
  float* curA    = (float*)carve((size_t)NPAD * 256 * 4);
  float* curB    = (float*)carve((size_t)NPAD * 256 * 4);
  float* hbuf    = (float*)carve((size_t)NPAD * 128 * 4);
  float* x4buf   = (float*)carve((size_t)NPAD * 128 * 4);
  GatBufs g;
  g.agg  = (float*)carve((size_t)NPAD * 128 * 4);
  g.als  = (float*)carve((size_t)NPAD * 4 * 4);
  g.ald  = (float*)carve((size_t)NPAD * 4 * 4);
  g.m    = (unsigned*)carve((size_t)NPAD * 4 * 4);
  g.ssum = (float*)carve((size_t)NPAD * 4 * 4);
  g.ebuf = (float*)carve((size_t)EE * 4 * 4);

  // ================= layer 1 (input dim 128) =================
  conv_f16_kernel<<<gdiv((long)NPAD * 128, 256), 256, 0, stream>>>(
      x, nullptr, Xh, (long)NN * 128, (long)NPAD * 128);
  // x1 = relu(x @ fc1_W + b) -> first[:, 0:128]
  run_conv_w(stream, fc1_W, Wt, 128);
  run_gemm(stream, Xh, Wt, fc1_b, first, 128, 256, 0, 1);
  // h = x @ g1_W
  run_conv_w(stream, g1_W, Wt, 128);
  run_gemm(stream, Xh, Wt, nullptr, hbuf, 128, 128, 0, 0);
  // x2 = relu(gat(...)) -> first[:, 128:256]
  run_gat(stream, hbuf, src, dst, g1_as, g1_ad, 4, 32, 5, g);
  gat_finish_kernel<<<gdiv((long)NPAD * 128, 256), 256, 0, stream>>>(g.agg, g1_b, first, 256, 128);

  // ================= mid layers (input dim 256) =================
  for (int i = 0; i < 2; ++i) {
    const float* li_a = (i == 0) ? first : curA;   // layer_input = cur (+ first for i>=1)
    const float* li_b = (i == 0) ? nullptr : first;
    float* curOut = (i == 0) ? curA : curB;
    conv_f16_kernel<<<gdiv((long)NPAD * 256, 256), 256, 0, stream>>>(
        li_a, li_b, Xh, (long)NN * 256, (long)NPAD * 256);
    run_conv_w(stream, mfc_W + (size_t)i * 256 * 128, Wt, 256);
    run_gemm(stream, Xh, Wt, mfc_b + (size_t)i * 128, curOut, 256, 256, 0, 1);
    run_conv_w(stream, mg_W + (size_t)i * 256 * 128, Wt, 256);
    run_gemm(stream, Xh, Wt, nullptr, hbuf, 256, 128, 0, 0);
    run_gat(stream, hbuf, src, dst, mg_as + (size_t)i * 128, mg_ad + (size_t)i * 128, 4, 32, 5, g);
    gat_finish_kernel<<<gdiv((long)NPAD * 128, 256), 256, 0, stream>>>(
        g.agg, mg_b + (size_t)i * 128, curOut, 256, 128);
  }

  // ================= final layer (input dim 256, single head) =================
  conv_f16_kernel<<<gdiv((long)NPAD * 256, 256), 256, 0, stream>>>(
      curB, nullptr, Xh, (long)NN * 256, (long)NPAD * 256);
  // x4 = relu(cur @ ffc_W + b)
  run_conv_w(stream, ffc_W, Wt, 256);
  run_gemm(stream, Xh, Wt, ffc_b, x4buf, 256, 128, 0, 1);
  // h = cur @ fg_W ; gat heads=1, C=128, concat=False (mean over 1 head == identity)
  run_conv_w(stream, fg_W, Wt, 256);
  run_gemm(stream, Xh, Wt, nullptr, hbuf, 256, 128, 0, 0);
  run_gat(stream, hbuf, src, dst, fg_as, fg_ad, 1, 128, 7, g);
  // out = x4 + relu(agg + fg_b)
  final_out_kernel<<<gdiv((long)NN * 128, 256), 256, 0, stream>>>(x4buf, g.agg, fg_b, out);
}
